// IsolatedFarFieldLongRangeV2_66881230733961
// MI455X (gfx1250) — compile-verified, matrix operand order checked by reading, not run on hardware
//
#include <hip/hip_runtime.h>
#include <hip/hip_bf16.h>
#include <math.h>

typedef __attribute__((ext_vector_type(16))) _Float16 v16h;
typedef __attribute__((ext_vector_type(8)))  _Float16 v8h;
typedef __attribute__((ext_vector_type(8)))  float    v8f;

#define CUTR   5.0f
#define EMAXR  40.0f

__device__ __forceinline__ float siluf(float x) { return x / (1.f + __expf(-x)); }

// ---- WMMA helpers (CDNA5 v_wmma_f32_16x16x32_f16, wave32) -----------------
// Fragment loader for row-major fp16 [16 rows x K] with leading dim `ld`.
// Works for A (rows = M) and for B when weights are stored transposed [N][Kpad].
// Per ISA 7.12.2 each lane needs two contiguous 8xf16 runs -> 2x ds_load_b128.
__device__ __forceinline__ v16h load_frag(const _Float16* base, int ld, int lane) {
  int row = lane & 15;
  int kb  = (lane & 16) ? 8 : 0;
  const _Float16* p = base + row * ld + kb;
  v8h lo = *(const v8h*)(p);
  v8h hi = *(const v8h*)(p + 16);
  return __builtin_shufflevector(lo, hi, 0, 1, 2, 3, 4, 5, 6, 7, 8, 9, 10, 11,
                                 12, 13, 14, 15);
}
__device__ __forceinline__ v8f wmma16(v16h a, v16h b, v8f c) {
  return __builtin_amdgcn_wmma_f32_16x16x32_f16(false, a, false, b, (short)0, c,
                                                false, false);
}
// Transpose-fill weights W[K x N] (global, f32) -> Wt[N x Kpad] (LDS, f16),
// zero-padding K..Kpad. One ds_store_b128 per 8 elements.
template <int K, int N, int Kpad>
__device__ __forceinline__ void fill_wt(const float* __restrict__ W,
                                        _Float16* Wt, int lane) {
  constexpr int CH = Kpad >> 3;
  for (int ch = lane; ch < N * CH; ch += 32) {
    int n = ch / CH;
    int k0 = (ch % CH) << 3;
    v8h t;
#pragma unroll
    for (int j = 0; j < 8; ++j) {
      int kk = k0 + j;
      t[j] = (_Float16)((kk < K) ? W[kk * N + n] : 0.f);
    }
    *(v8h*)(Wt + n * Kpad + k0) = t;
  }
}

// ===========================================================================
// Kernel 1: src = LN(x) -> MLP(128->64 silu ->16) -> LN.   1 wave / 16 atoms.
// ===========================================================================
__global__ __launch_bounds__(32) void src_kernel(
    const float* __restrict__ x, const float* __restrict__ lng0,
    const float* __restrict__ lnb0, const float* __restrict__ w1,
    const float* __restrict__ b1, const float* __restrict__ w2,
    const float* __restrict__ b2, const float* __restrict__ lng1,
    const float* __restrict__ lnb1, float* __restrict__ src) {
  __shared__ __align__(16) _Float16 xh[16 * 128];
  __shared__ __align__(16) _Float16 w1t[64 * 128];  // [n][k]
  __shared__ __align__(16) _Float16 h1h[16 * 64];
  __shared__ __align__(16) _Float16 w2t[16 * 64];   // [n][k]
  const int lane = threadIdx.x;
  const int tile = blockIdx.x;

  // LayerNorm rows of x: 2 lanes per row; row half cached in registers.
  {
    int row = lane >> 1, half = lane & 1;
    const float4* xr =
        (const float4*)(x + (size_t)(tile * 16 + row) * 128 + half * 64);
    float4 buf[16];
    float s = 0.f, s2 = 0.f;
#pragma unroll
    for (int c = 0; c < 16; ++c) {
      float4 v = xr[c];
      buf[c] = v;
      s += v.x + v.y + v.z + v.w;
      s2 += v.x * v.x + v.y * v.y + v.z * v.z + v.w * v.w;
    }
    s  += __shfl_xor(s, 1, 32);
    s2 += __shfl_xor(s2, 1, 32);
    float mean = s * (1.f / 128.f);
    float var  = s2 * (1.f / 128.f) - mean * mean;
    float inv  = rsqrtf(var + 1e-5f);
#pragma unroll
    for (int c8 = 0; c8 < 8; ++c8) {
      int cb = half * 64 + c8 * 8;
      float4 a = buf[2 * c8], b = buf[2 * c8 + 1];
      float4 g0 = ((const float4*)(lng0 + cb))[0];
      float4 g1 = ((const float4*)(lng0 + cb))[1];
      float4 o0 = ((const float4*)(lnb0 + cb))[0];
      float4 o1 = ((const float4*)(lnb0 + cb))[1];
      v8h t;
      t[0] = (_Float16)((a.x - mean) * inv * g0.x + o0.x);
      t[1] = (_Float16)((a.y - mean) * inv * g0.y + o0.y);
      t[2] = (_Float16)((a.z - mean) * inv * g0.z + o0.z);
      t[3] = (_Float16)((a.w - mean) * inv * g0.w + o0.w);
      t[4] = (_Float16)((b.x - mean) * inv * g1.x + o1.x);
      t[5] = (_Float16)((b.y - mean) * inv * g1.y + o1.y);
      t[6] = (_Float16)((b.z - mean) * inv * g1.z + o1.z);
      t[7] = (_Float16)((b.w - mean) * inv * g1.w + o1.w);
      *(v8h*)(xh + row * 128 + cb) = t;
    }
  }
  fill_wt<128, 64, 128>(w1, w1t, lane);
  __syncthreads();

  const int col = lane & 15;
  const int mh  = (lane & 16) ? 8 : 0;
  // GEMM1 [16,128]x[128,64] + silu
#pragma unroll
  for (int nt = 0; nt < 4; ++nt) {
    v8f acc = {0.f, 0.f, 0.f, 0.f, 0.f, 0.f, 0.f, 0.f};
#pragma unroll
    for (int kt = 0; kt < 4; ++kt) {
      v16h a = load_frag(xh + kt * 32, 128, lane);
      v16h b = load_frag(w1t + nt * 16 * 128 + kt * 32, 128, lane);
      acc = wmma16(a, b, acc);
    }
#pragma unroll
    for (int r = 0; r < 8; ++r) {
      float v = acc[r] + b1[nt * 16 + col];
      h1h[(r + mh) * 64 + nt * 16 + col] = (_Float16)siluf(v);
    }
  }
  fill_wt<64, 16, 64>(w2, w2t, lane);
  __syncthreads();

  // GEMM2 [16,64]x[64,16] + row LayerNorm (cross-lane)
  v8f acc = {0.f, 0.f, 0.f, 0.f, 0.f, 0.f, 0.f, 0.f};
#pragma unroll
  for (int kt = 0; kt < 2; ++kt) {
    v16h a = load_frag(h1h + kt * 32, 64, lane);
    v16h b = load_frag(w2t + kt * 32, 64, lane);
    acc = wmma16(a, b, acc);
  }
#pragma unroll
  for (int r = 0; r < 8; ++r) {
    float v = acc[r] + b2[col];
    float s = v;
    s += __shfl_xor(s, 1, 32); s += __shfl_xor(s, 2, 32);
    s += __shfl_xor(s, 4, 32); s += __shfl_xor(s, 8, 32);
    float mean = s * (1.f / 16.f);
    float dv = v - mean;
    float q = dv * dv;
    q += __shfl_xor(q, 1, 32); q += __shfl_xor(q, 2, 32);
    q += __shfl_xor(q, 4, 32); q += __shfl_xor(q, 8, 32);
    float inv = rsqrtf(q * (1.f / 16.f) + 1e-5f);
    src[(size_t)(tile * 16 + r + mh) * 16 + col] =
        dv * inv * lng1[col] + lnb1[col];
  }
}

// ===========================================================================
// Kernel 2: per-graph mean of src (512 contiguous atoms per graph)
// ===========================================================================
__global__ __launch_bounds__(256) void mean_kernel(const float* __restrict__ src,
                                                   float* __restrict__ meanv) {
  __shared__ float red[256];
  const int g = blockIdx.x, t = threadIdx.x;
  const int dim = t & 15;
  float s = 0.f;
  for (int a = (t >> 4); a < 512; a += 16)
    s += src[(size_t)(g * 512 + a) * 16 + dim];
  red[t] = s;
  __syncthreads();
  for (int step = 128; step >= 16; step >>= 1) {
    if (t < step) red[t] += red[t + step];
    __syncthreads();
  }
  if (t < 16) meanv[g * 16 + t] = red[t] * (1.f / 512.f);
}

// ===========================================================================
// Kernel 3: all-pairs shell statistics + per-pair RBF-gate MLP (VALU).
// 1 thread per i-atom, j-side pos/src resident in LDS (38 KB of 320 KB).
// ===========================================================================
#define SHELL_ACC(S)                                                           \
  do {                                                                         \
    const float* sp = sj + j * 16;                                             \
    _Pragma("unroll") for (int c = 0; c < 16; ++c) acc[S][c] += ker * sp[c];   \
    cnt[S] += 1.f; sr[S] += d; sr2[S] += d * d;                                \
  } while (0)

__global__ __launch_bounds__(128) void pair_kernel(
    const float* __restrict__ pos, const float* __restrict__ src,
    const float* __restrict__ meanv, const float* __restrict__ kscreen,
    const float* __restrict__ kgw1, const float* __restrict__ kgb1,
    const float* __restrict__ kgw2, const float* __restrict__ kgb2,
    float* __restrict__ shellF) {
  __shared__ __align__(16) float pj[512 * 3];
  __shared__ __align__(16) float sj[512 * 16];
  __shared__ float w1s[10 * 32];
  __shared__ float b1s[32];
  __shared__ float w2s[32];
  const int g = blockIdx.x >> 2;
  const int ib = blockIdx.x & 3;
  const int t = threadIdx.x;

  for (int idx4 = t; idx4 < 384; idx4 += 128)
    ((float4*)pj)[idx4] = ((const float4*)(pos + (size_t)g * 512 * 3))[idx4];
  {
    const float4 mg = ((const float4*)(meanv + g * 16))[t & 3];
    const float4* s4 = (const float4*)(src + (size_t)g * 512 * 16);
    for (int idx4 = t; idx4 < 2048; idx4 += 128) {
      float4 v = s4[idx4];
      v.x -= mg.x; v.y -= mg.y; v.z -= mg.z; v.w -= mg.w;
      ((float4*)sj)[idx4] = v;
    }
  }
  for (int idx = t; idx < 320; idx += 128) w1s[idx] = kgw1[idx];
  if (t < 32) { b1s[t] = kgb1[t]; w2s[t] = kgw2[t]; }
  __syncthreads();

  const int i = ib * 128 + t;
  const float pix = pj[i * 3 + 0], piy = pj[i * 3 + 1], piz = pj[i * 3 + 2];
  const float screening = log1pf(__expf(kscreen[0]));  // softplus
  const float kgb2v = kgb2[0];

  float acc[5][16], cnt[5], sr[5], sr2[5];
#pragma unroll
  for (int s = 0; s < 5; ++s) {
    cnt[s] = 0.f; sr[s] = 0.f; sr2[s] = 0.f;
#pragma unroll
    for (int c = 0; c < 16; ++c) acc[s][c] = 0.f;
  }

  for (int j = 0; j < 512; ++j) {
    float dx = pix - pj[j * 3 + 0];
    float dy = piy - pj[j * 3 + 1];
    float dz = piz - pj[j * 3 + 2];
    float d = sqrtf(dx * dx + dy * dy + dz * dz + 1e-12f);
    if (j == i || d < CUTR) continue;  // all shells start at r=5

    float base = __expf(-screening * d) / fmaxf(d, 1e-6f);
    // RBF gate: [d/5, d/40, exp(-(d-c_k)^2/25)], centers 5..40 step 5
    float gin[10];
    gin[0] = d * (1.f / CUTR);
    gin[1] = d * (1.f / EMAXR);
#pragma unroll
    for (int c = 0; c < 8; ++c) {
      float dd = d - (5.f + 5.f * (float)c);
      gin[2 + c] = __expf(-0.04f * dd * dd);
    }
    float z = kgb2v;
#pragma unroll 4
    for (int h = 0; h < 32; ++h) {
      float a = b1s[h];
#pragma unroll
      for (int k = 0; k < 10; ++k) a += gin[k] * w1s[k * 32 + h];
      z += siluf(a) * w2s[h];
    }
    float ker = base * (1.f + tanhf(z));

    if (d < 10.f)      SHELL_ACC(0);
    else if (d < 20.f) SHELL_ACC(1);
    else if (d < 40.f) SHELL_ACC(2);
    else if (d < 80.f) SHELL_ACC(3);
    else               SHELL_ACC(4);
  }

  float* orow = shellF + (size_t)(g * 512 + i) * 95;
#pragma unroll
  for (int s = 0; s < 5; ++s) {
    float denom = fmaxf(cnt[s], 1.f);
    float invd = 1.f / denom;
#pragma unroll
    for (int c = 0; c < 16; ++c) orow[s * 19 + c] = acc[s][c] * invd;
    orow[s * 19 + 16] = cnt[s];
    orow[s * 19 + 17] = sr[s] * invd;
    orow[s * 19 + 18] = sqrtf(sr2[s] * invd + 1e-12f);
  }
}

// ===========================================================================
// Kernel 4: shell->emb MLP, ein assembly, energy head, per-graph reduction.
// 1 wave per 16-atom tile; all dense layers on WMMA.
// ===========================================================================
__global__ __launch_bounds__(32) void head_kernel(
    const float* __restrict__ src, const float* __restrict__ meanv,
    const float* __restrict__ shellF, const float* __restrict__ se_ln_g,
    const float* __restrict__ se_ln_b, const float* __restrict__ se_w1,
    const float* __restrict__ se_b1, const float* __restrict__ se_w2,
    const float* __restrict__ se_b2, const float* __restrict__ eh_ln_g,
    const float* __restrict__ eh_ln_b, const float* __restrict__ eh_w1,
    const float* __restrict__ eh_b1, const float* __restrict__ eh_w2,
    const float* __restrict__ eh_b2, const float* __restrict__ far_gate,
    const float* __restrict__ energy_scale, float* __restrict__ out) {
  __shared__ __align__(16) _Float16 Sh[16 * 96];    // LN(shell), K pad 95->96
  __shared__ __align__(16) float    Sraw[16 * 95];  // raw shell (reused in ein)
  __shared__ __align__(16) _Float16 Wt[64 * 160];   // se_w1^T then eh_w1^T
  __shared__ __align__(16) _Float16 h1h[16 * 64];
  __shared__ __align__(16) _Float16 w2t[16 * 64];   // [n][k]
  __shared__ __align__(16) float    E[16 * 160];    // raw ein, K pad 159->160
  __shared__ __align__(16) _Float16 Eh[16 * 160];   // LN(ein)
  const int lane = threadIdx.x;
  const int tile = blockIdx.x;
  const int col = lane & 15;
  const int mh  = (lane & 16) ? 8 : 0;
  const int g   = (tile * 16) / 512;  // 512 % 16 == 0 -> uniform per tile

  for (int idx4 = lane; idx4 < 380; idx4 += 32)  // 16*95 = 1520 = 380*4
    ((float4*)Sraw)[idx4] =
        ((const float4*)(shellF + (size_t)tile * 16 * 95))[idx4];
  __syncthreads();
  if (lane < 16) {
    const float* r = Sraw + lane * 95;
    float s = 0.f, s2 = 0.f;
    for (int c = 0; c < 95; ++c) { s += r[c]; s2 += r[c] * r[c]; }
    float mean = s * (1.f / 95.f);
    float var = s2 * (1.f / 95.f) - mean * mean;
    float inv = rsqrtf(var + 1e-5f);
#pragma unroll
    for (int c8 = 0; c8 < 12; ++c8) {
      v8h t;
#pragma unroll
      for (int j = 0; j < 8; ++j) {
        int c = c8 * 8 + j;
        t[j] = (_Float16)((c < 95)
                              ? ((r[c] - mean) * inv * se_ln_g[c] + se_ln_b[c])
                              : 0.f);
      }
      *(v8h*)(Sh + lane * 96 + c8 * 8) = t;
    }
  }
  fill_wt<95, 64, 96>(se_w1, Wt, lane);
  __syncthreads();

  // shell-embed GEMM1 [16,96]x[96,64] + silu
#pragma unroll
  for (int nt = 0; nt < 4; ++nt) {
    v8f acc = {0.f, 0.f, 0.f, 0.f, 0.f, 0.f, 0.f, 0.f};
#pragma unroll
    for (int kt = 0; kt < 3; ++kt) {
      v16h a = load_frag(Sh + kt * 32, 96, lane);
      v16h b = load_frag(Wt + (nt * 16) * 96 + kt * 32, 96, lane);
      acc = wmma16(a, b, acc);
    }
#pragma unroll
    for (int r = 0; r < 8; ++r)
      h1h[(r + mh) * 64 + nt * 16 + col] =
          (_Float16)siluf(acc[r] + se_b1[nt * 16 + col]);
  }
  fill_wt<64, 16, 64>(se_w2, w2t, lane);
  __syncthreads();

  // GEMM2 [16,64]x[64,16] -> emb (C layout)
  v8f emb = {0.f, 0.f, 0.f, 0.f, 0.f, 0.f, 0.f, 0.f};
#pragma unroll
  for (int kt = 0; kt < 2; ++kt) {
    v16h a = load_frag(h1h + kt * 32, 64, lane);
    v16h b = load_frag(w2t + kt * 32, 64, lane);
    emb = wmma16(a, b, emb);
  }

  // assemble raw ein = [src_c, emb, src_c*emb, src_c-emb, shell]
#pragma unroll
  for (int r = 0; r < 8; ++r) {
    int m = r + mh;
    float sc = src[(size_t)(tile * 16 + m) * 16 + col] - meanv[g * 16 + col];
    float e = emb[r] + se_b2[col];
    E[m * 160 + col] = sc;
    E[m * 160 + 16 + col] = e;
    E[m * 160 + 32 + col] = sc * e;
    E[m * 160 + 48 + col] = sc - e;
  }
  for (int idx = lane; idx < 16 * 95; idx += 32) {
    int m = idx / 95, c = idx % 95;
    E[m * 160 + 64 + c] = Sraw[idx];
  }
  __syncthreads();

  // LN(ein): 2 lanes per row, contiguous halves [0,80) / [80,159)
  {
    int row = lane >> 1, half = lane & 1;
    int c0 = half * 80;
    int c1 = half ? 159 : 80;
    float s = 0.f, s2 = 0.f;
    for (int c = c0; c < c1; ++c) {
      float v = E[row * 160 + c];
      s += v; s2 += v * v;
    }
    s  += __shfl_xor(s, 1, 32);
    s2 += __shfl_xor(s2, 1, 32);
    float mean = s * (1.f / 159.f);
    float var = s2 * (1.f / 159.f) - mean * mean;
    float inv = rsqrtf(var + 1e-5f);
#pragma unroll
    for (int c8 = 0; c8 < 10; ++c8) {
      v8h t;
#pragma unroll
      for (int j = 0; j < 8; ++j) {
        int c = c0 + c8 * 8 + j;
        t[j] = (_Float16)((c < 159) ? ((E[row * 160 + c] - mean) * inv *
                                           eh_ln_g[c] + eh_ln_b[c])
                                    : 0.f);
      }
      *(v8h*)(Eh + row * 160 + c0 + c8 * 8) = t;
    }
  }
  fill_wt<159, 64, 160>(eh_w1, Wt, lane);
  __syncthreads();

  // energy head: GEMM [16,160]x[160,64], silu, dot with eh_w2 folded into lanes
  float p[8];
#pragma unroll
  for (int r = 0; r < 8; ++r) p[r] = 0.f;
#pragma unroll
  for (int nt = 0; nt < 4; ++nt) {
    v8f acc = {0.f, 0.f, 0.f, 0.f, 0.f, 0.f, 0.f, 0.f};
#pragma unroll
    for (int kt = 0; kt < 5; ++kt) {
      v16h a = load_frag(Eh + kt * 32, 160, lane);
      v16h b = load_frag(Wt + (nt * 16) * 160 + kt * 32, 160, lane);
      acc = wmma16(a, b, acc);
    }
    float w2v = eh_w2[nt * 16 + col];
#pragma unroll
    for (int r = 0; r < 8; ++r)
      p[r] += siluf(acc[r] + eh_b1[nt * 16 + col]) * w2v;
  }
  const float scale = tanhf(far_gate[0]) * __expf(energy_scale[0]);
  const float eb2 = eh_b2[0];
#pragma unroll
  for (int r = 0; r < 8; ++r) {
    float s = p[r];
    s += __shfl_xor(s, 1, 32); s += __shfl_xor(s, 2, 32);
    s += __shfl_xor(s, 4, 32); s += __shfl_xor(s, 8, 32);
    if (col == 0) atomicAdd(&out[g], (s + eb2) * scale);
  }
}

// ===========================================================================
extern "C" void kernel_launch(void* const* d_in, const int* in_sizes, int n_in,
                              void* d_out, int out_size, void* d_ws,
                              size_t ws_size, hipStream_t stream) {
  const float* x        = (const float*)d_in[0];
  const float* pos      = (const float*)d_in[1];
  // d_in[2] = batch (contiguous slabs; not needed), d_in[3] = num_graphs
  const float* in_ln_g  = (const float*)d_in[4];
  const float* in_ln_b  = (const float*)d_in[5];
  const float* src_w1   = (const float*)d_in[6];
  const float* src_b1   = (const float*)d_in[7];
  const float* src_w2   = (const float*)d_in[8];
  const float* src_b2   = (const float*)d_in[9];
  const float* src_ln_g = (const float*)d_in[10];
  const float* src_ln_b = (const float*)d_in[11];
  const float* se_ln_g  = (const float*)d_in[12];
  const float* se_ln_b  = (const float*)d_in[13];
  const float* se_w1    = (const float*)d_in[14];
  const float* se_b1    = (const float*)d_in[15];
  const float* se_w2    = (const float*)d_in[16];
  const float* se_b2    = (const float*)d_in[17];
  const float* eh_ln_g  = (const float*)d_in[18];
  const float* eh_ln_b  = (const float*)d_in[19];
  const float* eh_w1    = (const float*)d_in[20];
  const float* eh_b1    = (const float*)d_in[21];
  const float* eh_w2    = (const float*)d_in[22];
  const float* eh_b2    = (const float*)d_in[23];
  const float* k_screen = (const float*)d_in[24];
  const float* kg_w1    = (const float*)d_in[25];
  const float* kg_b1    = (const float*)d_in[26];
  const float* kg_w2    = (const float*)d_in[27];
  const float* kg_b2    = (const float*)d_in[28];
  const float* far_gate = (const float*)d_in[29];
  const float* e_scale  = (const float*)d_in[30];

  const int M  = in_sizes[0] / 128;  // 8192 atoms total
  const int nG = out_size;           // 16 graphs (output is per-graph energy)

  float* ws      = (float*)d_ws;
  float* w_src   = ws;                        // M*16 floats
  float* w_mean  = w_src + (size_t)M * 16;    // nG*16 floats
  float* w_shell = w_mean + (size_t)nG * 16;  // M*95 floats

  hipMemsetAsync(d_out, 0, (size_t)out_size * sizeof(float), stream);

  src_kernel<<<M / 16, 32, 0, stream>>>(x, in_ln_g, in_ln_b, src_w1, src_b1,
                                        src_w2, src_b2, src_ln_g, src_ln_b,
                                        w_src);
  mean_kernel<<<nG, 256, 0, stream>>>(w_src, w_mean);
  pair_kernel<<<nG * 4, 128, 0, stream>>>(pos, w_src, w_mean, k_screen, kg_w1,
                                          kg_b1, kg_w2, kg_b2, w_shell);
  head_kernel<<<M / 16, 32, 0, stream>>>(
      w_src, w_mean, w_shell, se_ln_g, se_ln_b, se_w1, se_b1, se_w2, se_b2,
      eh_ln_g, eh_ln_b, eh_w1, eh_b1, eh_w2, eh_b2, far_gate, e_scale,
      (float*)d_out);
}
